// GNN_41927470744091
// MI455X (gfx1250) — matrix operand order repaired and checked by reading, not compiled, not run on hardware
//
#include <hip/hip_runtime.h>
#include <stdint.h>

#define N_NODES  50000
#define N_EDGES  600000
#define D_IN     128
#define H1       128
#define H2       256
#define N_GRAPHS 1024

typedef __bf16 v16bf __attribute__((ext_vector_type(16)));
typedef float  v8f   __attribute__((ext_vector_type(8)));

union FragAB { v16bf v; uint4 q[2]; };

__device__ __forceinline__ unsigned short f2bf(float f) {
    unsigned int u = __float_as_uint(f);
    u += 0x7fffu + ((u >> 16) & 1u);          // round-to-nearest-even
    return (unsigned short)(u >> 16);
}
__device__ __forceinline__ float bf2f(unsigned int h) {
    return __uint_as_float((h & 0xffffu) << 16);
}

// ---- fp32 -> bf16 bulk convert (8 elems / thread) -------------------------
__global__ void k_f32_to_bf16(const float* __restrict__ in,
                              unsigned short* __restrict__ out, int n8) {
    int i = blockIdx.x * blockDim.x + threadIdx.x;
    if (i >= n8) return;
    const float4* p = (const float4*)in + 2 * (size_t)i;
    float4 a = p[0], b = p[1];
    uint4 o;
    o.x = f2bf(a.x) | ((unsigned)f2bf(a.y) << 16);
    o.y = f2bf(a.z) | ((unsigned)f2bf(a.w) << 16);
    o.z = f2bf(b.x) | ((unsigned)f2bf(b.y) << 16);
    o.w = f2bf(b.z) | ((unsigned)f2bf(b.w) << 16);
    ((uint4*)out)[i] = o;
}

// ---- weight [K,N] fp32 -> transposed bf16 [N,K] (K-contiguous B frags) ----
__global__ void k_wt_bf16(const float* __restrict__ w,
                          unsigned short* __restrict__ wt, int K, int N) {
    int i = blockIdx.x * blockDim.x + threadIdx.x;
    if (i >= K * N) return;
    int k = i / N, n = i % N;
    wt[(size_t)n * K + k] = f2bf(w[i]);
}

// ---- in-degree counts -----------------------------------------------------
__global__ void k_count(const int* __restrict__ dst, float* __restrict__ cnt, int E) {
    int e = blockIdx.x * blockDim.x + threadIdx.x;
    if (e < E) atomicAdd(&cnt[dst[e]], 1.0f);
}

// ---- per-edge scatter-add: one wave per edge, bf16 gather, f32 atomics ----
__global__ void k_scatter(const unsigned short* __restrict__ feat,
                          const int* __restrict__ src, const int* __restrict__ dst,
                          float* __restrict__ acc, int E) {
    int t = blockIdx.x * blockDim.x + threadIdx.x;
    int e = t >> 5, lane = t & 31;
    if (e >= E) return;
    int s = src[e], d = dst[e];
    uint2 u = *((const uint2*)(feat + (size_t)s * 128) + lane);  // 4 bf16 per lane
    float* q = acc + (size_t)d * 128 + lane * 4;
    atomicAdd(q + 0, bf2f(u.x));
    atomicAdd(q + 1, bf2f(u.x >> 16));
    atomicAdd(q + 2, bf2f(u.y));
    atomicAdd(q + 3, bf2f(u.y >> 16));
}

// ---- mean = sum / max(cnt,1), packed to bf16 (8 elems / thread) -----------
__global__ void k_mean_bf16(const float* __restrict__ sum, const float* __restrict__ cnt,
                            unsigned short* __restrict__ mean) {
    int i = blockIdx.x * blockDim.x + threadIdx.x;   // N_NODES*16 exact
    int n = i >> 4, j = i & 15;
    float rinv = 1.0f / fmaxf(cnt[n], 1.0f);
    const float4* p = (const float4*)(sum + (size_t)n * 128 + j * 8);
    float4 a = p[0], b = p[1];
    uint4 o;
    o.x = f2bf(a.x * rinv) | ((unsigned)f2bf(a.y * rinv) << 16);
    o.y = f2bf(a.z * rinv) | ((unsigned)f2bf(a.w * rinv) << 16);
    o.z = f2bf(b.x * rinv) | ((unsigned)f2bf(b.y * rinv) << 16);
    o.w = f2bf(b.z * rinv) | ((unsigned)f2bf(b.w * rinv) << 16);
    *((uint4*)(mean + (size_t)n * 128) + j) = o;
}

// ---- fused dual-GEMM + bias + row L2-normalize (+ReLU), WMMA bf16 ---------
// out[m, :] = normalize(mean@Wl + bias + x@Wr); block = 16 rows, wave = 16 cols
template <int NT, bool RELU>
__global__ __launch_bounds__(NT * 32) void k_gemm(
    const unsigned short* __restrict__ Am,   // mean  bf16 [N,128]
    const unsigned short* __restrict__ Ax,   // x / h bf16 [N,128]
    const unsigned short* __restrict__ WL,   // Wl^T  bf16 [NT*16,128]
    const unsigned short* __restrict__ WR,   // Wr^T  bf16 [NT*16,128]
    const float* __restrict__ bias,
    unsigned short* __restrict__ out)        // bf16 [N, NT*16]
{
    const int NC = NT * 16;
    int wave = threadIdx.x >> 5;
    int lane = threadIdx.x & 31;
    int lm = lane & 15, hf = lane >> 4;
    int mblock = blockIdx.x * 16;

    const unsigned short* arM = Am + (size_t)(mblock + lm) * 128;
    const unsigned short* arX = Ax + (size_t)(mblock + lm) * 128;
    int ncol = wave * 16 + lm;
    const unsigned short* brL = WL + (size_t)ncol * 128;
    const unsigned short* brR = WR + (size_t)ncol * 128;

    v8f c = {};
#pragma unroll
    for (int kc = 0; kc < 4; ++kc) {
        int kb = kc * 32;
        int s0 = kb + hf * 8;          // A/B: K {0-7 | 8-15} of chunk
        int s1 = kb + 16 + hf * 8;     // A/B: K {16-23 | 24-31} of chunk
        FragAB a, b;
        a.q[0] = *(const uint4*)(arM + s0);
        a.q[1] = *(const uint4*)(arM + s1);
        b.q[0] = *(const uint4*)(brL + s0);
        b.q[1] = *(const uint4*)(brL + s1);
        c = __builtin_amdgcn_wmma_f32_16x16x32_bf16(false, a.v, false, b.v,
                                                    (short)0, c, false, false);
        a.q[0] = *(const uint4*)(arX + s0);
        a.q[1] = *(const uint4*)(arX + s1);
        b.q[0] = *(const uint4*)(brR + s0);
        b.q[1] = *(const uint4*)(brR + s1);
        c = __builtin_amdgcn_wmma_f32_16x16x32_bf16(false, a.v, false, b.v,
                                                    (short)0, c, false, false);
    }

    float bv = bias[ncol];
    float cv[8];
    __shared__ float rowsq[16];
    if (threadIdx.x < 16) rowsq[threadIdx.x] = 0.0f;
    __syncthreads();
#pragma unroll
    for (int r = 0; r < 8; ++r) {              // C layout: VGPR r -> M = r + 8*hf
        cv[r] = c[r] + bv;
        atomicAdd(&rowsq[r + 8 * hf], cv[r] * cv[r]);
    }
    __syncthreads();
#pragma unroll
    for (int r = 0; r < 8; ++r) {
        int lr = r + 8 * hf;
        float inv = 1.0f / fmaxf(sqrtf(rowsq[lr]), 1e-12f);
        float v = cv[r] * inv;
        if (RELU) v = fmaxf(v, 0.0f);
        out[(size_t)(mblock + lr) * NC + ncol] = f2bf(v);
    }
}

// ---- segment mean-pool over sorted batch + linear + sigmoid ---------------
__device__ __forceinline__ int lbound(const int* a, int n, int key) {
    int lo = 0, hi = n;
    while (lo < hi) { int mid = (lo + hi) >> 1; if (a[mid] < key) lo = mid + 1; else hi = mid; }
    return lo;
}

__global__ void k_pool(const unsigned short* __restrict__ out2,
                       const int* __restrict__ batch,
                       const float* __restrict__ wlin, const float* __restrict__ blin,
                       float* __restrict__ out) {
    int g = blockIdx.x, cidx = threadIdx.x;
    int lo = lbound(batch, N_NODES, g);
    int hi = lbound(batch, N_NODES, g + 1);
    float s = 0.0f;
    for (int n = lo; n < hi; ++n) s += bf2f(out2[(size_t)n * H2 + cidx]);
    float denom = fmaxf((float)(hi - lo), 1.0f);
    float t = (s / denom) * wlin[cidx];
    __shared__ float red[H2];
    red[cidx] = t;
    __syncthreads();
    for (int st = H2 / 2; st > 0; st >>= 1) {
        if (cidx < st) red[cidx] += red[cidx + st];
        __syncthreads();
    }
    if (cidx == 0) out[g] = 1.0f / (1.0f + expf(-(red[0] + blin[0])));
}

extern "C" void kernel_launch(void* const* d_in, const int* in_sizes, int n_in,
                              void* d_out, int out_size, void* d_ws, size_t ws_size,
                              hipStream_t stream) {
    (void)in_sizes; (void)n_in; (void)out_size; (void)ws_size;
    const float* x    = (const float*)d_in[0];
    const int*   ei   = (const int*)d_in[1];
    const int*   batch= (const int*)d_in[3];
    const float* w1l  = (const float*)d_in[4];
    const float* b1   = (const float*)d_in[5];
    const float* w1r  = (const float*)d_in[6];
    const float* w2l  = (const float*)d_in[7];
    const float* b2   = (const float*)d_in[8];
    const float* w2r  = (const float*)d_in[9];
    const float* wlin = (const float*)d_in[10];
    const float* blin = (const float*)d_in[11];
    const int* src = ei;
    const int* dst = ei + N_EDGES;

    char* ws = (char*)d_ws;
    size_t off = 0;
    auto take = [&](size_t bytes) -> char* {
        char* p = ws + off;
        off += (bytes + 255) & ~(size_t)255;
        return p;
    };
    float*          cnt   = (float*)take((size_t)N_NODES * 4);
    float*          acc   = (float*)take((size_t)N_NODES * 128 * 4);
    unsigned short* xb    = (unsigned short*)take((size_t)N_NODES * 128 * 2);
    unsigned short* meanb = (unsigned short*)take((size_t)N_NODES * 128 * 2);
    unsigned short* hb    = (unsigned short*)take((size_t)N_NODES * 128 * 2);
    unsigned short* out2  = (unsigned short*)take((size_t)N_NODES * 256 * 2);
    unsigned short* wt1l  = (unsigned short*)take(128 * 128 * 2);
    unsigned short* wt1r  = (unsigned short*)take(128 * 128 * 2);
    unsigned short* wt2l  = (unsigned short*)take(128 * 256 * 2);
    unsigned short* wt2r  = (unsigned short*)take(128 * 256 * 2);

    hipMemsetAsync(cnt, 0, (size_t)N_NODES * 4, stream);
    hipMemsetAsync(acc, 0, (size_t)N_NODES * 128 * 4, stream);

    k_f32_to_bf16<<<3125, 256, 0, stream>>>(x, xb, N_NODES * 128 / 8);
    k_wt_bf16<<<64, 256, 0, stream>>>(w1l, wt1l, 128, 128);
    k_wt_bf16<<<64, 256, 0, stream>>>(w1r, wt1r, 128, 128);
    k_wt_bf16<<<128, 256, 0, stream>>>(w2l, wt2l, 128, 256);
    k_wt_bf16<<<128, 256, 0, stream>>>(w2r, wt2r, 128, 256);
    k_count<<<(N_EDGES + 255) / 256, 256, 0, stream>>>(dst, cnt, N_EDGES);

    // Layer 1
    k_scatter<<<N_EDGES / 8, 256, 0, stream>>>(xb, src, dst, acc, N_EDGES);
    k_mean_bf16<<<3125, 256, 0, stream>>>(acc, cnt, meanb);
    k_gemm<8, true><<<N_NODES / 16, 256, 0, stream>>>(meanb, xb, wt1l, wt1r, b1, hb);

    // Layer 2
    hipMemsetAsync(acc, 0, (size_t)N_NODES * 128 * 4, stream);
    k_scatter<<<N_EDGES / 8, 256, 0, stream>>>(hb, src, dst, acc, N_EDGES);
    k_mean_bf16<<<3125, 256, 0, stream>>>(acc, cnt, meanb);
    k_gemm<16, false><<<N_NODES / 16, 512, 0, stream>>>(meanb, hb, wt2l, wt2r, b2, out2);

    // Pool + classifier
    k_pool<<<N_GRAPHS, 256, 0, stream>>>(out2, batch, wlin, blin, (float*)d_out);
}
